// FeatureMixer_1288490189005
// MI455X (gfx1250) — matrix operand order repaired and checked by reading, not compiled, not run on hardware
//
#include <hip/hip_runtime.h>
#include <hip/hip_bf16.h>

// ---- vector types -----------------------------------------------------------
typedef __attribute__((ext_vector_type(16))) __bf16 v16bf;
typedef __attribute__((ext_vector_type(8)))  __bf16 v8bf;
typedef __attribute__((ext_vector_type(8)))  float  v8f;
typedef __attribute__((ext_vector_type(4)))  float  v4f;
typedef __attribute__((ext_vector_type(4)))  unsigned short v4u16;

#define K_DIM     768
#define LDS_S     776        // padded bf16 row stride: 1552B -> lanes land 4 banks apart
#define ROWS_BLK  32         // rows per block (two 16-row WMMA tiles per wave)
#define NEG_SLOPE 0.01f
#define LN_EPS    1e-5f

__device__ __forceinline__ unsigned short f32_to_bf16_rne(float f) {
    unsigned int u = __float_as_uint(f);
    u += 0x7FFFu + ((u >> 16) & 1u);     // round-to-nearest-even
    return (unsigned short)(u >> 16);
}

// W[k][n] (fp32, row-major KxNC) -> WT[n][k] (bf16, row-major NCxK)
__global__ void convert_transpose_bf16(const float* __restrict__ W,
                                       unsigned short* __restrict__ WT,
                                       int ncol) {
    int idx = blockIdx.x * 256 + threadIdx.x;
    if (idx >= ncol * K_DIM) return;
    int n = idx / K_DIM;
    int k = idx - n * K_DIM;
    WT[idx] = f32_to_bf16_rne(W[(size_t)k * ncol + n]);
}

// A fragment from LDS, 16-bit A 16x32 layout (CDNA5 ISA §7.12.2):
// lanes 0-15 (row m=p): elems 0..7 = K+0..7, elems 8..15 = K+16..23; lanes 16-31: +8
__device__ __forceinline__ v16bf load_a_frag(const unsigned short* sA,
                                             int m, int k0, int hi) {
    const unsigned short* pa = sA + m * LDS_S + k0 + hi * 8;
    v8bf lo = *(const v8bf*)pa;
    v8bf h2 = *(const v8bf*)(pa + 16);
    v16bf a;
#pragma unroll
    for (int i = 0; i < 8; ++i) { a[i] = lo[i]; a[i + 8] = h2[i]; }
    return a;
}

// B fragment from global bf16 WT[n][k]: lane holds column n; lanes 0-15 elems 0..7 = K+0..7,
// elems 8..15 = K+16..23; lanes 16-31: +8 (matches A; two b128 loads)
__device__ __forceinline__ v16bf load_b_frag(const unsigned short* __restrict__ WT,
                                             int n, int k0, int hi) {
    const unsigned short* pb = WT + (size_t)n * K_DIM + k0 + hi * 8;
    v8bf lo = *(const v8bf*)pb;
    v8bf h2 = *(const v8bf*)(pb + 16);
    v16bf b;
#pragma unroll
    for (int i = 0; i < 8; ++i) { b[i] = lo[i]; b[i + 8] = h2[i]; }
    return b;
}

// One block: 32 rows x (CT*128) cols. 8 waves; wave w owns cols [w*CT*16, ...) and
// BOTH 16-row tiles (each B fragment feeds 2 WMMAs). B is software-pipelined.
template <int CT>
__global__ void __launch_bounds__(256)
ln_gemm_lrelu(const float* __restrict__ X,
              const float* __restrict__ gamma,
              const float* __restrict__ beta,
              const unsigned short* __restrict__ WT,  // [NC][K_DIM] bf16
              const float* __restrict__ bias,
              float* __restrict__ Y) {
    constexpr int NC = CT * 128;
    __shared__ unsigned short sA[ROWS_BLK * LDS_S];

    const int  tid  = threadIdx.x;
    const int  wave = tid >> 5;
    const int  lane = tid & 31;
    const int  hi   = lane >> 4;     // half-wave id
    const int  p    = lane & 15;     // lane within half
    const long rowBase = (long)blockIdx.x * ROWS_BLK;

    // ---------------- Stage 1: LayerNorm 32 rows -> bf16 A tiles in LDS ------
#pragma unroll
    for (int it = 0; it < 2; ++it) {
        const int    row = wave * 4 + it * 2 + hi;     // 8 waves * 2 iters * 2 halves
        const float* xr  = X + (rowBase + row) * (long)K_DIM;
        v4f   xv[12];
        float sum = 0.f, ss = 0.f;
#pragma unroll
        for (int c = 0; c < 12; ++c) {                 // 16 lanes * 12 * float4 = 768
            v4f v = ((const v4f*)xr)[c * 16 + p];
            xv[c] = v;
            sum += v.x + v.y + v.z + v.w;
            ss  += v.x * v.x + v.y * v.y + v.z * v.z + v.w * v.w;
        }
#pragma unroll
        for (int off = 8; off >= 1; off >>= 1) {       // reduce across 16-lane half
            sum += __shfl_xor(sum, off, 16);
            ss  += __shfl_xor(ss,  off, 16);
        }
        const float mu  = sum * (1.0f / K_DIM);
        const float var = ss * (1.0f / K_DIM) - mu * mu;
        const float rs  = rsqrtf(var + LN_EPS);
#pragma unroll
        for (int c = 0; c < 12; ++c) {
            const int col = c * 64 + p * 4;
            v4f g = *(const v4f*)(gamma + col);
            v4f b = *(const v4f*)(beta  + col);
            v4f y = (xv[c] - mu) * rs * g + b;
            v4u16 o;
            o.x = f32_to_bf16_rne(y.x);
            o.y = f32_to_bf16_rne(y.y);
            o.z = f32_to_bf16_rne(y.z);
            o.w = f32_to_bf16_rne(y.w);
            *(v4u16*)&sA[row * LDS_S + col] = o;       // 8B-aligned ds_store
        }
    }
    __syncthreads();

    // ---------------- Stage 2: pipelined WMMA GEMM, K = 768 ------------------
    const int colBase = wave * (CT * 16);
    v8f acc0[CT] = {};     // rows 0..15
    v8f acc1[CT] = {};     // rows 16..31

    // prologue: first A pair (LDS) + first B tile in flight
    v16bf a0 = load_a_frag(sA, p,      0, hi);
    v16bf a1 = load_a_frag(sA, p + 16, 0, hi);
    v16bf bc = load_b_frag(WT, colBase + p, 0, hi);

    for (int k0 = 0; k0 < K_DIM; k0 += 32) {
        const int kn = (k0 + 32 < K_DIM) ? (k0 + 32) : 0;  // wrap: harmless dead loads
        v16bf a0n, a1n;
#pragma unroll
        for (int t = 0; t < CT; ++t) {
            v16bf bn;
            if (t == CT - 1) {
                // prefetch next k-step: A pair from LDS + first B tile
                a0n = load_a_frag(sA, p,      kn, hi);
                a1n = load_a_frag(sA, p + 16, kn, hi);
                bn  = load_b_frag(WT, colBase + p, kn, hi);
            } else {
                bn  = load_b_frag(WT, colBase + (t + 1) * 16 + p, k0, hi);
            }
            acc0[t] = __builtin_amdgcn_wmma_f32_16x16x32_bf16(
                false, a0, false, bc, (short)0, acc0[t], false, false);
            acc1[t] = __builtin_amdgcn_wmma_f32_16x16x32_bf16(
                false, a1, false, bc, (short)0, acc1[t], false, false);
            bc = bn;
        }
        a0 = a0n;
        a1 = a1n;
    }

    // ---------------- Stage 3: bias + LeakyReLU epilogue ---------------------
#pragma unroll
    for (int t = 0; t < CT; ++t) {
        const int   n  = colBase + t * 16 + p;
        const float bb = bias[n];
#pragma unroll
        for (int r = 0; r < 8; ++r) {
            const int m = hi * 8 + r;                  // C/D layout: VGPR r, half hi
            float v0 = acc0[t][r] + bb;
            v0 = (v0 >= 0.f) ? v0 : NEG_SLOPE * v0;
            Y[(rowBase + m) * (long)NC + n] = v0;
            float v1 = acc1[t][r] + bb;
            v1 = (v1 >= 0.f) ? v1 : NEG_SLOPE * v1;
            Y[(rowBase + 16 + m) * (long)NC + n] = v1;
        }
    }
}

extern "C" void kernel_launch(void* const* d_in, const int* in_sizes, int n_in,
                              void* d_out, int out_size, void* d_ws, size_t ws_size,
                              hipStream_t stream) {
    const float* h         = (const float*)d_in[0];
    const float* z         = (const float*)d_in[1];
    const float* ctx_gamma = (const float*)d_in[2];
    const float* ctx_beta  = (const float*)d_in[3];
    const float* ctx_W     = (const float*)d_in[4];   // [768][768]
    const float* ctx_b     = (const float*)d_in[5];
    const float* vec_gamma = (const float*)d_in[6];
    const float* vec_beta  = (const float*)d_in[7];
    const float* vec_W     = (const float*)d_in[8];   // [768][512]
    const float* vec_b     = (const float*)d_in[9];

    const int N = in_sizes[0] / K_DIM;                // 65536

    unsigned short* ctxWT = (unsigned short*)d_ws;             // 768*768 bf16
    unsigned short* vecWT = ctxWT + 768 * 768;                 // 512*768 bf16

    float* s0 = (float*)d_out;                 // [N][768]
    float* s  = s0 + (size_t)N * 768;          // [N][512]

    convert_transpose_bf16<<<(768 * 768 + 255) / 256, 256, 0, stream>>>(ctx_W, ctxWT, 768);
    convert_transpose_bf16<<<(768 * 512 + 255) / 256, 256, 0, stream>>>(vec_W, vecWT, 512);

    ln_gemm_lrelu<6><<<N / ROWS_BLK, 256, 0, stream>>>(z, ctx_gamma, ctx_beta, ctxWT, ctx_b, s0);
    ln_gemm_lrelu<4><<<N / ROWS_BLK, 256, 0, stream>>>(h, vec_gamma, vec_beta, vecWT, vec_b, s);
}
